// AtomicEmbedder_18236431139167
// MI455X (gfx1250) — matrix operand order repaired and checked by reading, not compiled, not run on hardware
//
#include <hip/hip_runtime.h>
#include <hip/hip_bf16.h>

typedef __attribute__((ext_vector_type(16))) _Float16 v16h;
typedef __attribute__((ext_vector_type(8)))  float    v8f;
typedef __attribute__((ext_vector_type(4)))  unsigned v4u;
typedef __attribute__((ext_vector_type(8)))  int      v8i;
typedef __attribute__((ext_vector_type(4)))  int      v4i;

#define B_   8192
#define F_   16
#define V_   1000
#define D_   512
#define OUT_ 1024
#define L_   4
#define H_   4
#define DH_  128
#define S_   17
#define FF_  2048
#define MT_  (B_ * S_)   // 139264 token rows

// ---------------- WMMA fragment helpers (CDNA5 16x16x32 f16 layouts) ----------------

__device__ __forceinline__ v16h frag_pack(uint4 a, uint4 b) {
  union { uint4 u[2]; v16h v; } t;
  t.u[0] = a; t.u[1] = b;
  return t.v;
}

// A fragment (16x32, row-major source, leading dim lda in halves).
__device__ __forceinline__ v16h load_a_frag(const _Float16* __restrict__ Ak0, int lda) {
  int lane = threadIdx.x & 31;
  const _Float16* p = Ak0 + (size_t)(lane & 15) * lda + ((lane >> 4) << 3);
  return frag_pack(*(const uint4*)p, *(const uint4*)(p + 16));
}

// B fragment (32x16) from transposed weights WT[n][k] (row-major, leading dim ldk).
__device__ __forceinline__ v16h load_b_frag(const _Float16* __restrict__ WTn0, int ldk, int k0) {
  int lane = threadIdx.x & 31;
  const _Float16* p = WTn0 + (size_t)(lane & 15) * ldk + k0 + ((lane >> 4) << 3);
  return frag_pack(*(const uint4*)p, *(const uint4*)(p + 16));
}

__device__ __forceinline__ v8f wmma16(v16h a, v16h b, v8f c) {
  return __builtin_amdgcn_wmma_f32_16x16x32_f16(false, a, false, b, (short)0, c, false, false);
}

__device__ __forceinline__ float wave_sum(float x) {
#pragma unroll
  for (int o = 16; o > 0; o >>= 1) x += __shfl_xor(x, o, 32);
  return x;
}

__device__ __forceinline__ float gelu_exact(float x) {
  return 0.5f * x * (1.0f + erff(x * 0.70710678118654752f));
}

// ---------------- CDNA5 async global->LDS (ASYNCcnt-tracked, 16B granules) ----------------

__device__ __forceinline__ unsigned lds_off_of(const void* p) {
  // generic LDS pointer: addr[31:0] is the LDS byte address (ISA aperture mapping)
  return (unsigned)(size_t)p;
}

__device__ __forceinline__ void async_ld_b128(unsigned lds_off, const void* gaddr) {
  asm volatile("global_load_async_to_lds_b128 %0, %1, off"
               :: "v"(lds_off), "v"(gaddr) : "memory");
}

__device__ __forceinline__ void async_wait0() {
  asm volatile("s_wait_asynccnt 0x0" ::: "memory");
}

// ---------------- CDNA5 Tensor Data Mover: 2D tile -> LDS with row padding ----------------
// D# per CDNA5 ISA: group0 = {count/lds_addr/global_addr/type},
// group1 = {data_size=2B, pad_enable, pad_interval=256dw (1 row), pad_amount=4dw (16B),
//           tensor_dim0=512, tensor_dim1=rows, tile_dim0=512, tile_dim1=rows, dim0_stride}.
// LDS row pitch becomes 512 halves + 8 pad halves = 520 (matches compute-side layout).

__device__ __forceinline__ void tdm_load_2d(unsigned lds_byte_addr, const void* gaddr,
                                            unsigned rows, unsigned row_stride_elems) {
  unsigned long long ga = (unsigned long long)(size_t)gaddr;
  v4u g0;
  g0[0] = 1u;                                               // count=1, user descriptor
  g0[1] = lds_byte_addr;                                    // lds_addr
  g0[2] = (unsigned)(ga & 0xffffffffu);                     // global_addr[31:0]
  g0[3] = (unsigned)((ga >> 32) & 0x01ffffffu) | (2u << 30);  // global_addr[56:32], type=2
  v8i g1;
  g1[0] = (int)((1u << 16) | (1u << 20) | (7u << 22) | (3u << 25));
  //            data_size=2B | pad_enable | pad_interval=256dw | pad_amount=4dw
  g1[1] = (int)(512u << 16);         // tensor_dim0[15:0] (bits 63:48)
  g1[2] = (int)(rows << 16);         // tensor_dim1[15:0] (bits 95:80)
  g1[3] = (int)(512u << 16);         // tile_dim0 = 512 elems (bits 127:112)
  g1[4] = (int)rows;                 // tile_dim1 (bits 143:128)
  g1[5] = (int)row_stride_elems;     // tensor_dim0_stride[31:0] (bits 191:160)
  g1[6] = 0;
  g1[7] = 0;
  v4i zz = {0, 0, 0, 0};
#if __clang_major__ >= 23
  v8i z8 = {0, 0, 0, 0, 0, 0, 0, 0};
  __builtin_amdgcn_tensor_load_to_lds(g0, g1, zz, zz, z8, 0);
#else
  __builtin_amdgcn_tensor_load_to_lds(g0, g1, zz, zz, 0);
#endif
}

__device__ __forceinline__ void tensor_wait0() {
  __builtin_amdgcn_s_wait_tensorcnt((unsigned short)0);
}

// stage ROWS x 512 halves via TDM (global row stride in halves, LDS pitch 520)
// Issued by wave 0 only; tensorcnt-wait + barrier makes it visible block-wide.
__device__ __forceinline__ void stage_tile_tdm(const _Float16* __restrict__ gsrc,
                                               _Float16* lds_tile, unsigned rows,
                                               unsigned row_stride_elems) {
  if (threadIdx.x < 32) {
    tdm_load_2d(lds_off_of(lds_tile), gsrc, rows, row_stride_elems);
    tensor_wait0();
  }
  __syncthreads();
}

// ---------------- Weight convert + transpose (f32 [K,N] -> f16 [N,K]) ----------------

__global__ void k_cvt_transpose(const float* __restrict__ src, _Float16* __restrict__ dst,
                                int K, int N) {
  __shared__ float tile[32][33];
  int tx = threadIdx.x & 31, ty = threadIdx.x >> 5;
  int n0 = blockIdx.x * 32, k0 = blockIdx.y * 32;
#pragma unroll
  for (int i = 0; i < 4; ++i) {
    int k = k0 + ty + i * 8, n = n0 + tx;
    if (k < K && n < N) tile[ty + i * 8][tx] = src[(size_t)k * N + n];
  }
  __syncthreads();
#pragma unroll
  for (int i = 0; i < 4; ++i) {
    int n = n0 + ty + i * 8, k = k0 + tx;
    if (k < K && n < N) dst[(size_t)n * K + k] = (_Float16)tile[tx][ty + i * 8];
  }
}

// ---------------- Embedding gather + LayerNorm -> h16 ----------------

__global__ void k_embed_ln(const int* __restrict__ feats, const unsigned char* __restrict__ has,
                           const float* __restrict__ cls, const float* __restrict__ tables,
                           const float* __restrict__ fill, const float* __restrict__ g,
                           const float* __restrict__ bt, _Float16* __restrict__ h16) {
  int wave = threadIdx.x >> 5, lane = threadIdx.x & 31;
  size_t row = (size_t)blockIdx.x * 8 + wave;
  if (row >= (size_t)MT_) return;
  int b = (int)(row / S_), s = (int)(row % S_);

  float x[16];
  if (s == 0) {
#pragma unroll
    for (int i = 0; i < 16; ++i) x[i] = cls[lane + i * 32];
  } else {
    int f = s - 1;
    int idx = feats[b * F_ + f];
    const float* srcp = (has[b * F_ + f] != 0)
                            ? tables + ((size_t)f * V_ + (size_t)idx) * D_
                            : fill + (size_t)f * D_;
#pragma unroll
    for (int i = 0; i < 16; ++i) x[i] = srcp[lane + i * 32];
  }

  float s1 = 0.0f;
#pragma unroll
  for (int i = 0; i < 16; ++i) s1 += x[i];
  float mu = wave_sum(s1) * (1.0f / D_);
  float s2 = 0.0f;
#pragma unroll
  for (int i = 0; i < 16; ++i) { float d = x[i] - mu; s2 += d * d; }
  float rs = rsqrtf(wave_sum(s2) * (1.0f / D_) + 1e-7f);

#pragma unroll
  for (int i = 0; i < 16; ++i) {
    int c = lane + i * 32;
    h16[row * D_ + c] = (_Float16)((x[i] - mu) * rs * g[c] + bt[c]);
  }
}

// ---------------- QKV GEMM (M=64 blocks): [MT,512] x [512,1536] + bias ----------------

__global__ void k_gemm_qkv(const _Float16* __restrict__ h16, const _Float16* __restrict__ wqkvT,
                           const float* __restrict__ bq, const float* __restrict__ bk,
                           const float* __restrict__ bv, _Float16* __restrict__ qkv16) {
  __shared__ _Float16 atile[64 * 520];
  int tid = threadIdx.x, wave = tid >> 5, lane = tid & 31;
  size_t m0 = (size_t)blockIdx.x * 64;

  stage_tile_tdm(h16 + m0 * D_, atile, 64, D_);

  int n0 = blockIdx.y * 128 + wave * 16;
  v8f z = {0.f, 0.f, 0.f, 0.f, 0.f, 0.f, 0.f, 0.f};
  v8f acc[4] = {z, z, z, z};
#pragma unroll
  for (int kk = 0; kk < 16; ++kk) {
    v16h b = load_b_frag(wqkvT + (size_t)n0 * D_, D_, kk * 32);
    v16h a[4];
#pragma unroll
    for (int mt = 0; mt < 4; ++mt)
      a[mt] = load_a_frag(atile + mt * 16 * 520 + kk * 32, 520);
#pragma unroll
    for (int mt = 0; mt < 4; ++mt) acc[mt] = wmma16(a[mt], b, acc[mt]);
  }

  int mb = (lane >> 4) * 8, nl = lane & 15;
  int n = n0 + nl;
  float bias = (n < 512) ? bq[n] : (n < 1024 ? bk[n - 512] : bv[n - 1024]);
#pragma unroll
  for (int mt = 0; mt < 4; ++mt)
#pragma unroll
    for (int r = 0; r < 8; ++r)
      qkv16[(m0 + mt * 16 + mb + r) * (3 * D_) + n] = (_Float16)(acc[mt][r] + bias);
}

// ---------------- Attention (S=17): one block per batch, VALU + async staging ----------------

__global__ void k_attn(const _Float16* __restrict__ qkv16, _Float16* __restrict__ ctx16) {
  __shared__ _Float16 smem[S_ * 3 * D_];  // 52 KB
  int tid = threadIdx.x;
  const _Float16* gbase = qkv16 + (size_t)blockIdx.x * S_ * (3 * D_);
  // 17*1536 halves = 3264 16B-chunks, async to LDS
#pragma unroll
  for (int j = 0; j < 13; ++j) {
    int c = tid + j * 256;
    if (c < (S_ * 3 * D_) / 8)
      async_ld_b128(lds_off_of(smem + c * 8), gbase + (size_t)c * 8);
  }
  async_wait0();
  __syncthreads();

  int wave = tid >> 5, lane = tid & 31;
  const float scale = 0.08838834764831845f;  // 1/sqrt(128)

  for (int pair = wave; pair < H_ * S_; pair += 8) {
    int head = pair / S_, sq = pair - head * S_;
    int dbase = head * DH_ + lane * 4;
    const _Float16* qp = &smem[sq * (3 * D_) + dbase];
    float q0 = (float)qp[0], q1 = (float)qp[1], q2 = (float)qp[2], q3 = (float)qp[3];

    float sc[S_];
#pragma unroll
    for (int key = 0; key < S_; ++key) {
      const _Float16* kp = &smem[key * (3 * D_) + D_ + dbase];
      float p = q0 * (float)kp[0] + q1 * (float)kp[1] + q2 * (float)kp[2] + q3 * (float)kp[3];
      sc[key] = wave_sum(p) * scale;
    }
    float mx = sc[0];
#pragma unroll
    for (int k = 1; k < S_; ++k) mx = fmaxf(mx, sc[k]);
    float den = 0.0f;
#pragma unroll
    for (int k = 0; k < S_; ++k) { sc[k] = __expf(sc[k] - mx); den += sc[k]; }
    float inv = 1.0f / den;

    float a0 = 0.f, a1 = 0.f, a2 = 0.f, a3 = 0.f;
#pragma unroll
    for (int key = 0; key < S_; ++key) {
      const _Float16* vp = &smem[key * (3 * D_) + 2 * D_ + dbase];
      float w = sc[key] * inv;
      a0 += w * (float)vp[0]; a1 += w * (float)vp[1];
      a2 += w * (float)vp[2]; a3 += w * (float)vp[3];
    }
    _Float16* op = ctx16 + ((size_t)blockIdx.x * S_ + sq) * D_ + dbase;
    op[0] = (_Float16)a0; op[1] = (_Float16)a1; op[2] = (_Float16)a2; op[3] = (_Float16)a3;
  }
}

// ---------------- Attn-out GEMM (M=64) + bias + residual + LayerNorm ----------------

__global__ void k_attnout_ln(const _Float16* __restrict__ ctx16, const _Float16* __restrict__ woT,
                             const float* __restrict__ bo, const float* __restrict__ g,
                             const float* __restrict__ bt, _Float16* __restrict__ h16) {
  __shared__ _Float16 atile[64 * 520];
  __shared__ float s_sum[8][64], s_sq[8][64];
  __shared__ float s_mu[64], s_rs[64];
  int tid = threadIdx.x, wave = tid >> 5, lane = tid & 31;
  size_t m0 = (size_t)blockIdx.x * 64;

  stage_tile_tdm(ctx16 + m0 * D_, atile, 64, D_);

  v8f z = {0.f, 0.f, 0.f, 0.f, 0.f, 0.f, 0.f, 0.f};
  v8f acc[4][4];
#pragma unroll
  for (int mt = 0; mt < 4; ++mt)
#pragma unroll
    for (int t = 0; t < 4; ++t) acc[mt][t] = z;

#pragma unroll
  for (int kk = 0; kk < 16; ++kk) {
    v16h bf[4];
#pragma unroll
    for (int t = 0; t < 4; ++t)
      bf[t] = load_b_frag(woT + (size_t)(wave * 64 + t * 16) * D_, D_, kk * 32);
    v16h a[4];
#pragma unroll
    for (int mt = 0; mt < 4; ++mt)
      a[mt] = load_a_frag(atile + mt * 16 * 520 + kk * 32, 520);
#pragma unroll
    for (int mt = 0; mt < 4; ++mt)
#pragma unroll
      for (int t = 0; t < 4; ++t) acc[mt][t] = wmma16(a[mt], bf[t], acc[mt][t]);
  }

  int mb = (lane >> 4) * 8, nl = lane & 15;
  // pass 1: per-wave row partial sums of x = acc + bo + residual
#pragma unroll
  for (int mt = 0; mt < 4; ++mt) {
#pragma unroll
    for (int r = 0; r < 8; ++r) {
      int row = mt * 16 + mb + r;
      float ps = 0.f, pq = 0.f;
#pragma unroll
      for (int t = 0; t < 4; ++t) {
        int n = wave * 64 + t * 16 + nl;
        float x = acc[mt][t][r] + bo[n] + (float)h16[(m0 + row) * D_ + n];
        ps += x; pq += x * x;
      }
#pragma unroll
      for (int o = 1; o < 16; o <<= 1) { ps += __shfl_xor(ps, o, 32); pq += __shfl_xor(pq, o, 32); }
      if (nl == 0) { s_sum[wave][row] = ps; s_sq[wave][row] = pq; }
    }
  }
  __syncthreads();
  if (tid < 64) {
    float s = 0.f, q = 0.f;
#pragma unroll
    for (int w = 0; w < 8; ++w) { s += s_sum[w][tid]; q += s_sq[w][tid]; }
    float mu = s * (1.0f / D_);
    s_mu[tid] = mu;
    s_rs[tid] = rsqrtf(q * (1.0f / D_) - mu * mu + 1e-7f);
  }
  __syncthreads();
  // pass 2: normalize + write
#pragma unroll
  for (int mt = 0; mt < 4; ++mt) {
#pragma unroll
    for (int r = 0; r < 8; ++r) {
      int row = mt * 16 + mb + r;
      float mu = s_mu[row], rs = s_rs[row];
#pragma unroll
      for (int t = 0; t < 4; ++t) {
        int n = wave * 64 + t * 16 + nl;
        float x = acc[mt][t][r] + bo[n] + (float)h16[(m0 + row) * D_ + n];
        h16[(m0 + row) * D_ + n] = (_Float16)((x - mu) * rs * g[n] + bt[n]);
      }
    }
  }
}

// ---------------- Fully fused FFN (M=64): GEMM1 + GELU + GEMM2 + residual + LN ----------------

__global__ void k_ffn_ln(_Float16* __restrict__ h16, const _Float16* __restrict__ w1T,
                         const float* __restrict__ b1, const _Float16* __restrict__ w2T,
                         const float* __restrict__ b2, const float* __restrict__ g,
                         const float* __restrict__ bt) {
  __shared__ _Float16 atile[64 * 520];  // 66.6 KB
  __shared__ _Float16 utile[64 * 136];  // 17.4 KB
  __shared__ float s_sum[8][64], s_sq[8][64];
  __shared__ float s_mu[64], s_rs[64];
  int tid = threadIdx.x, wave = tid >> 5, lane = tid & 31;
  size_t m0 = (size_t)blockIdx.x * 64;
  int mb = (lane >> 4) * 8, nl = lane & 15;

  stage_tile_tdm(h16 + m0 * D_, atile, 64, D_);

  v8f z = {0.f, 0.f, 0.f, 0.f, 0.f, 0.f, 0.f, 0.f};
  v8f acco[4][4];
#pragma unroll
  for (int mt = 0; mt < 4; ++mt)
#pragma unroll
    for (int t = 0; t < 4; ++t) acco[mt][t] = z;

  for (int ch = 0; ch < 16; ++ch) {  // 16 chunks of 128 FFN columns
    // Phase 1: u = gelu(h @ W1 + b1) for this wave's 16 chunk-columns, 4 M-tiles
    int un0 = ch * 128 + wave * 16;
    v8f u[4] = {z, z, z, z};
#pragma unroll
    for (int kk = 0; kk < 16; ++kk) {
      v16h bf = load_b_frag(w1T + (size_t)un0 * D_, D_, kk * 32);
      v16h a[4];
#pragma unroll
      for (int mt = 0; mt < 4; ++mt)
        a[mt] = load_a_frag(atile + mt * 16 * 520 + kk * 32, 520);
#pragma unroll
      for (int mt = 0; mt < 4; ++mt) u[mt] = wmma16(a[mt], bf, u[mt]);
    }
    float bias = b1[un0 + nl];
    int ucol = wave * 16 + nl;
#pragma unroll
    for (int mt = 0; mt < 4; ++mt)
#pragma unroll
      for (int r = 0; r < 8; ++r)
        utile[(mt * 16 + mb + r) * 136 + ucol] = (_Float16)gelu_exact(u[mt][r] + bias);
    __syncthreads();

    // Phase 2: out += u_chunk @ W2[chunk,:]
#pragma unroll
    for (int k2 = 0; k2 < 4; ++k2) {
      v16h bf2[4];
#pragma unroll
      for (int t = 0; t < 4; ++t)
        bf2[t] = load_b_frag(w2T + (size_t)(wave * 64 + t * 16) * FF_, FF_, ch * 128 + k2 * 32);
      v16h a2[4];
#pragma unroll
      for (int mt = 0; mt < 4; ++mt)
        a2[mt] = load_a_frag(utile + mt * 16 * 136 + k2 * 32, 136);
#pragma unroll
      for (int mt = 0; mt < 4; ++mt)
#pragma unroll
        for (int t = 0; t < 4; ++t) acco[mt][t] = wmma16(a2[mt], bf2[t], acco[mt][t]);
    }
    __syncthreads();
  }

  // Epilogue: bias + residual + LayerNorm (register-resident two-pass)
#pragma unroll
  for (int mt = 0; mt < 4; ++mt) {
#pragma unroll
    for (int r = 0; r < 8; ++r) {
      int row = mt * 16 + mb + r;
      float ps = 0.f, pq = 0.f;
#pragma unroll
      for (int t = 0; t < 4; ++t) {
        int n = wave * 64 + t * 16 + nl;
        float x = acco[mt][t][r] + b2[n] + (float)atile[row * 520 + n];
        ps += x; pq += x * x;
      }
#pragma unroll
      for (int o = 1; o < 16; o <<= 1) { ps += __shfl_xor(ps, o, 32); pq += __shfl_xor(pq, o, 32); }
      if (nl == 0) { s_sum[wave][row] = ps; s_sq[wave][row] = pq; }
    }
  }
  __syncthreads();
  if (tid < 64) {
    float s = 0.f, q = 0.f;
#pragma unroll
    for (int w = 0; w < 8; ++w) { s += s_sum[w][tid]; q += s_sq[w][tid]; }
    float mu = s * (1.0f / D_);
    s_mu[tid] = mu;
    s_rs[tid] = rsqrtf(q * (1.0f / D_) - mu * mu + 1e-7f);
  }
  __syncthreads();
#pragma unroll
  for (int mt = 0; mt < 4; ++mt) {
#pragma unroll
    for (int r = 0; r < 8; ++r) {
      int row = mt * 16 + mb + r;
      float mu = s_mu[row], rs = s_rs[row];
#pragma unroll
      for (int t = 0; t < 4; ++t) {
        int n = wave * 64 + t * 16 + nl;
        float x = acco[mt][t][r] + b2[n] + (float)atile[row * 520 + n];
        h16[(m0 + row) * D_ + n] = (_Float16)((x - mu) * rs * g[n] + bt[n]);
      }
    }
  }
}

// ---------------- Pooler: tanh(h[:,0] @ pool_W + pool_b) -> f32 out ----------------

__global__ void k_pool(const _Float16* __restrict__ h16, const _Float16* __restrict__ pWT,
                       const float* __restrict__ pb, float* __restrict__ out) {
  __shared__ _Float16 atile[16 * 520];
  int tid = threadIdx.x, wave = tid >> 5, lane = tid & 31;
  size_t b0 = (size_t)blockIdx.x * 16;

  // CLS rows: strided 2D gather (row stride S_*D_ halves) — TDM descriptor handles it
  stage_tile_tdm(h16 + b0 * S_ * D_, atile, 16, S_ * D_);

  v8f z = {0.f, 0.f, 0.f, 0.f, 0.f, 0.f, 0.f, 0.f};
  v8f acc[8] = {z, z, z, z, z, z, z, z};
#pragma unroll
  for (int kk = 0; kk < 16; ++kk) {
    v16h a = load_a_frag(atile + kk * 32, 520);
#pragma unroll
    for (int t = 0; t < 8; ++t) {
      int n0 = wave * 128 + t * 16;
      v16h b = load_b_frag(pWT + (size_t)n0 * D_, D_, kk * 32);
      acc[t] = wmma16(a, b, acc[t]);
    }
  }

  int mb = (lane >> 4) * 8, nl = lane & 15;
#pragma unroll
  for (int t = 0; t < 8; ++t) {
    int n = wave * 128 + t * 16 + nl;
#pragma unroll
    for (int r = 0; r < 8; ++r)
      out[(b0 + mb + r) * OUT_ + n] = tanhf(acc[t][r] + pb[n]);
  }
}

// ---------------- Host orchestration ----------------

extern "C" void kernel_launch(void* const* d_in, const int* in_sizes, int n_in,
                              void* d_out, int out_size, void* d_ws, size_t ws_size,
                              hipStream_t stream) {
  const int* feats           = (const int*)d_in[0];
  const unsigned char* has   = (const unsigned char*)d_in[1];
  const float* cls           = (const float*)d_in[2];
  const float* tables        = (const float*)d_in[3];
  const float* fill          = (const float*)d_in[4];
  const float* eg            = (const float*)d_in[5];
  const float* eb            = (const float*)d_in[6];
  const float* Wq            = (const float*)d_in[7];
  const float* bq            = (const float*)d_in[8];
  const float* Wk            = (const float*)d_in[9];
  const float* bk            = (const float*)d_in[10];
  const float* Wv            = (const float*)d_in[11];
  const float* bv            = (const float*)d_in[12];
  const float* Wo            = (const float*)d_in[13];
  const float* bo            = (const float*)d_in[14];
  const float* ag            = (const float*)d_in[15];
  const float* ab            = (const float*)d_in[16];
  const float* W1            = (const float*)d_in[17];
  const float* b1            = (const float*)d_in[18];
  const float* W2            = (const float*)d_in[19];
  const float* b2            = (const float*)d_in[20];
  const float* fg            = (const float*)d_in[21];
  const float* fb            = (const float*)d_in[22];
  const float* pW            = (const float*)d_in[23];
  const float* pb            = (const float*)d_in[24];

  char* ws = (char*)d_ws;
  const size_t HPL = 3145728;  // f16 elems per layer weight block (qkvT+oT+w1T+w2T)
  _Float16* wf16 = (_Float16*)ws;
  _Float16* poolWT = wf16 + 4 * HPL;
  size_t off = (4 * HPL + (size_t)OUT_ * D_) * sizeof(_Float16);
  off = (off + 255) & ~(size_t)255;
  _Float16* h16 = (_Float16*)(ws + off);   off += (size_t)MT_ * D_ * 2;
  _Float16* ctx16 = (_Float16*)(ws + off); off += (size_t)MT_ * D_ * 2;
  _Float16* qkv16 = (_Float16*)(ws + off); off += (size_t)MT_ * 3 * D_ * 2;
  (void)ws_size; (void)in_sizes; (void)n_in; (void)out_size;

  // Weight convert + transpose (all f32 [K,N] -> f16 [N,K])
  for (int l = 0; l < L_; ++l) {
    _Float16* lb    = wf16 + (size_t)l * HPL;
    _Float16* wqkvT = lb;
    _Float16* woT   = lb + 786432;
    _Float16* w1T   = woT + 262144;
    _Float16* w2T   = w1T + 1048576;
    k_cvt_transpose<<<dim3(16, 16), 256, 0, stream>>>(Wq + (size_t)l * D_ * D_, wqkvT, D_, D_);
    k_cvt_transpose<<<dim3(16, 16), 256, 0, stream>>>(Wk + (size_t)l * D_ * D_, wqkvT + 512 * 512, D_, D_);
    k_cvt_transpose<<<dim3(16, 16), 256, 0, stream>>>(Wv + (size_t)l * D_ * D_, wqkvT + 1024 * 512, D_, D_);
    k_cvt_transpose<<<dim3(16, 16), 256, 0, stream>>>(Wo + (size_t)l * D_ * D_, woT, D_, D_);
    k_cvt_transpose<<<dim3(64, 16), 256, 0, stream>>>(W1 + (size_t)l * D_ * FF_, w1T, D_, FF_);
    k_cvt_transpose<<<dim3(16, 64), 256, 0, stream>>>(W2 + (size_t)l * FF_ * D_, w2T, FF_, D_);
  }
  k_cvt_transpose<<<dim3(32, 16), 256, 0, stream>>>(pW, poolWT, D_, OUT_);

  // Embedding + LN
  k_embed_ln<<<MT_ / 8, 256, 0, stream>>>(feats, has, cls, tables, fill, eg, eb, h16);

  // Transformer layers
  for (int l = 0; l < L_; ++l) {
    _Float16* lb    = wf16 + (size_t)l * HPL;
    _Float16* wqkvT = lb;
    _Float16* woT   = lb + 786432;
    _Float16* w1T   = woT + 262144;
    _Float16* w2T   = w1T + 1048576;
    k_gemm_qkv<<<dim3(MT_ / 64, 12), 256, 0, stream>>>(h16, wqkvT, bq + l * D_, bk + l * D_,
                                                       bv + l * D_, qkv16);
    k_attn<<<B_, 256, 0, stream>>>(qkv16, ctx16);
    k_attnout_ln<<<MT_ / 64, 256, 0, stream>>>(ctx16, woT, bo + l * D_, ag + l * D_,
                                               ab + l * D_, h16);
    k_ffn_ln<<<MT_ / 64, 256, 0, stream>>>(h16, w1T, b1 + (size_t)l * FF_, w2T, b2 + l * D_,
                                           fg + l * D_, fb + l * D_);
  }

  // Pooler
  k_pool<<<B_ / 16, 256, 0, stream>>>(h16, poolWT, pb, (float*)d_out);
}
